// RandNLAGQALayer_24223615549564
// MI455X (gfx1250) — matrix operand order repaired and checked by reading, not compiled, not run on hardware
//
#include <hip/hip_runtime.h>
#include <hip/hip_bf16.h>
#include <math.h>

typedef __attribute__((ext_vector_type(16))) _Float16 v16h;
typedef __attribute__((ext_vector_type(8)))  _Float16 v8h;
typedef __attribute__((ext_vector_type(8)))  float    v8f;

#define S_LEN 4096
#define HIDN  2048
#define NH    16
#define NKV   4
#define HD    128
#define SK    640
#define KTOP  2048
#define T_TOT (KTOP + SK)   // 2688
#define EPSV  1e-6f

// ---------------------------------------------------------------------------
// WMMA fragment loaders (wave32, 16x16x32 f16 shapes; layouts per CDNA5 ISA)
// A (16x32, MxK): lane<16 holds row M=lane, K {k0..k0+7, k0+16..k0+23};
//                 lane>=16 holds row M=lane-16, K {k0+8..15, k0+24..31}.
// B (32x16, KxN): lane<16 holds col N=lane, K {k0..k0+7, k0+16..k0+23};
//                 lane>=16 holds col N=lane-16, K {k0+8..15, k0+24..31}.
// ---------------------------------------------------------------------------
__device__ __forceinline__ v16h load_a_frag(const _Float16* base, int row0, int k0, int ld) {
  int lane = threadIdx.x & 31;
  const _Float16* p = base + (size_t)(row0 + (lane & 15)) * ld + k0 + ((lane & 16) ? 8 : 0);
  v8h lo = *(const v8h*)p;
  v8h hi = *(const v8h*)(p + 16);
  v16h f;
#pragma unroll
  for (int i = 0; i < 8; ++i) { f[i] = lo[i]; f[i + 8] = hi[i]; }
  return f;
}

__device__ __forceinline__ v16h load_b_frag(const _Float16* base, int col0, int k0, int ld) {
  int lane = threadIdx.x & 31;
  const _Float16* p = base + (size_t)(col0 + (lane & 15)) * ld + k0 + ((lane & 16) ? 8 : 0);
  v8h lo = *(const v8h*)p;
  v8h hi = *(const v8h*)(p + 16);
  v16h f;
#pragma unroll
  for (int i = 0; i < 8; ++i) { f[i] = lo[i]; f[i + 8] = hi[i]; }
  return f;
}

// ---------------------------------------------------------------------------
// Elementwise f32 -> f16
// ---------------------------------------------------------------------------
__global__ void __launch_bounds__(256) f32_to_f16_kernel(const float* __restrict__ in,
                                                         _Float16* __restrict__ out, int n) {
  int i = blockIdx.x * 256 + threadIdx.x;
  if (i < n) out[i] = (_Float16)in[i];
}

// ---------------------------------------------------------------------------
// Transpose + convert: W[K][N] f32  ->  Wt[N][K] f16
// ---------------------------------------------------------------------------
__global__ void __launch_bounds__(256) transpose_w_kernel(const float* __restrict__ W,
                                                          _Float16* __restrict__ Wt,
                                                          int K, int N) {
  __shared__ float tile[32][33];
  int k0 = blockIdx.y * 32, n0 = blockIdx.x * 32;
  int tx = threadIdx.x, ty = threadIdx.y;          // block (32,8)
  for (int r = ty; r < 32; r += 8) tile[r][tx] = W[(size_t)(k0 + r) * N + n0 + tx];
  __syncthreads();
  for (int r = ty; r < 32; r += 8) Wt[(size_t)(n0 + r) * K + k0 + tx] = (_Float16)tile[tx][r];
}

// ---------------------------------------------------------------------------
// WMMA GEMM with 64x64 register tile per wave:
//   C[M,N] = A[M,K](f16) * Bt[N,K](f16), f32 accumulate.
// Block = 128 threads = 4 waves; block tile = 64 rows x 256 cols.
// Per k-step: 4 A-frags + 4 B-frags feed 16 WMMAs (0.5KB loaded per WMMA,
// 16 independent MMAs in flight to hide L2 latency).
// ---------------------------------------------------------------------------
__global__ void __launch_bounds__(128) gemm_f16_wmma(const _Float16* __restrict__ A,
                                                     const _Float16* __restrict__ Bt,
                                                     float* __restrict__ C,
                                                     int M, int N, int K) {
  int wave = threadIdx.x >> 5;
  int m0 = blockIdx.y * 64;
  int n0 = blockIdx.x * 256 + wave * 64;
  v8f acc[4][4] = {};
  for (int k0 = 0; k0 < K; k0 += 32) {
    __builtin_prefetch(A + (size_t)m0 * K + k0 + 256, 0, 1);
    __builtin_prefetch(Bt + (size_t)n0 * K + k0 + 256, 0, 1);
    v16h a[4], b[4];
#pragma unroll
    for (int i = 0; i < 4; ++i) a[i] = load_a_frag(A, m0 + i * 16, k0, K);
#pragma unroll
    for (int j = 0; j < 4; ++j) b[j] = load_b_frag(Bt, n0 + j * 16, k0, K);
#pragma unroll
    for (int i = 0; i < 4; ++i)
#pragma unroll
      for (int j = 0; j < 4; ++j)
        acc[i][j] = __builtin_amdgcn_wmma_f32_16x16x32_f16(false, a[i], false, b[j],
                                                           (short)0, acc[i][j], false, false);
  }
  int lane = threadIdx.x & 31;
  int col = lane & 15;
#pragma unroll
  for (int i = 0; i < 4; ++i)
#pragma unroll
    for (int j = 0; j < 4; ++j)
#pragma unroll
      for (int e = 0; e < 8; ++e) {
        int row = (lane & 16) ? e + 8 : e;
        C[(size_t)(m0 + i * 16 + row) * N + n0 + j * 16 + col] = acc[i][j][e];
      }
}

// ---------------------------------------------------------------------------
// Importance MLP: logit[s] = tanh(x[s]@W1+b1)@W2 + b2 - log(S/SKETCH)
// ---------------------------------------------------------------------------
__global__ void __launch_bounds__(64) mlp_logits_kernel(const float* __restrict__ x,
                                                        const float* __restrict__ W1,
                                                        const float* __restrict__ b1,
                                                        const float* __restrict__ W2,
                                                        const float* __restrict__ b2,
                                                        float* __restrict__ logits) {
  __shared__ float xs[HIDN];
  __shared__ float red[64];
  int s = blockIdx.x, j = threadIdx.x;
  for (int i = j; i < HIDN; i += 64) xs[i] = x[(size_t)s * HIDN + i];
  __syncthreads();
  float h = b1[j];
  for (int i = 0; i < HIDN; ++i) h = fmaf(xs[i], W1[(size_t)i * 64 + j], h);
  h = tanhf(h);
  red[j] = h * W2[j];
  __syncthreads();
  for (int off = 32; off > 0; off >>= 1) {
    if (j < off) red[j] += red[j + off];
    __syncthreads();
  }
  if (j == 0) logits[s] = red[0] + b2[0] - logf((float)S_LEN / (float)SK);
}

// ---------------------------------------------------------------------------
// Exact top-K select (single block): bitwise threshold search + prefix scans.
// idx[] = selected positions ascending; istop[] = membership flags.
// ---------------------------------------------------------------------------
__global__ void __launch_bounds__(1024) topk_select_kernel(const float* __restrict__ logits,
                                                           int* __restrict__ idx,
                                                           int* __restrict__ istop) {
  __shared__ unsigned keys[S_LEN];
  __shared__ int sc[1024];
  __shared__ int cnt;
  __shared__ unsigned thr;
  int tid = threadIdx.x;
  for (int i = tid; i < S_LEN; i += 1024) {
    unsigned b = __float_as_uint(logits[i]);
    keys[i] = (b & 0x80000000u) ? ~b : (b | 0x80000000u);   // order-preserving map
  }
  if (tid == 0) thr = 0u;
  __syncthreads();
  for (int b = 31; b >= 0; --b) {
    unsigned cand = thr | (1u << b);
    if (tid == 0) cnt = 0;
    __syncthreads();
    int c = 0;
    for (int i = tid; i < S_LEN; i += 1024) c += (keys[i] >= cand) ? 1 : 0;
    atomicAdd(&cnt, c);
    __syncthreads();
    if (tid == 0 && cnt >= KTOP) thr = cand;
    __syncthreads();
  }
  unsigned T = thr;                       // largest v with count(key>=v) >= KTOP
  if (tid == 0) cnt = 0;
  __syncthreads();
  {
    int c = 0;
    for (int i = tid; i < S_LEN; i += 1024) c += (keys[i] > T) ? 1 : 0;
    atomicAdd(&cnt, c);
  }
  __syncthreads();
  int needEq = KTOP - cnt;                // ties admitted in position order
  int base = tid * 4;
  int eqc = 0;
#pragma unroll
  for (int u = 0; u < 4; ++u) eqc += (keys[base + u] == T) ? 1 : 0;
  sc[tid] = eqc;
  __syncthreads();
  for (int off = 1; off < 1024; off <<= 1) {
    int v = sc[tid];
    int add = (tid >= off) ? sc[tid - off] : 0;
    __syncthreads();
    sc[tid] = v + add;
    __syncthreads();
  }
  int eqExcl = sc[tid] - eqc;
  int selFlag[4];
  int selc = 0, run = 0;
#pragma unroll
  for (int u = 0; u < 4; ++u) {
    unsigned kk = keys[base + u];
    int sel = 0;
    if (kk > T) sel = 1;
    else if (kk == T) { sel = (eqExcl + run < needEq) ? 1 : 0; run++; }
    selFlag[u] = sel;
    selc += sel;
    istop[base + u] = sel;
  }
  __syncthreads();
  sc[tid] = selc;
  __syncthreads();
  for (int off = 1; off < 1024; off <<= 1) {
    int v = sc[tid];
    int add = (tid >= off) ? sc[tid - off] : 0;
    __syncthreads();
    sc[tid] = v + add;
    __syncthreads();
  }
  int pos = sc[tid] - selc;
#pragma unroll
  for (int u = 0; u < 4; ++u)
    if (selFlag[u]) idx[pos++] = base + u;
}

// ---------------------------------------------------------------------------
// rest_w[s] = sigmoid(logit[s]) * !is_topk[s]
// ---------------------------------------------------------------------------
__global__ void __launch_bounds__(256) restw_kernel(const float* __restrict__ logits,
                                                    const int* __restrict__ istop,
                                                    float* __restrict__ restw) {
  int i = blockIdx.x * 256 + threadIdx.x;
  if (i >= S_LEN) return;
  float iw = 1.0f / (1.0f + __expf(-logits[i]));
  restw[i] = istop[i] ? 0.0f : iw;
}

// ---------------------------------------------------------------------------
// RMSNorm + RoPE on Q; writes qh[h][s][d] f16. One wave per (s,h) row.
// ---------------------------------------------------------------------------
__global__ void __launch_bounds__(128) rms_rope_q_kernel(const float* __restrict__ q,
                                                         const int* __restrict__ pos_ids,
                                                         const float* __restrict__ qw,
                                                         _Float16* __restrict__ qh) {
  int wave = threadIdx.x >> 5, lane = threadIdx.x & 31;
  int rowid = blockIdx.x * 4 + wave;           // s*NH + h
  int s = rowid / NH, h = rowid % NH;
  const float* src = q + (size_t)s * (NH * HD) + (size_t)h * HD;
  float v[4];
  float ss = 0.0f;
#pragma unroll
  for (int u = 0; u < 4; ++u) { v[u] = src[lane * 4 + u]; ss += v[u] * v[u]; }
  for (int off = 16; off > 0; off >>= 1) ss += __shfl_xor(ss, off);
  float rs = rsqrtf(ss / (float)HD + EPSV);
  float pos = (float)pos_ids[s];
  _Float16* dst = qh + ((size_t)h * S_LEN + s) * HD;
#pragma unroll
  for (int u = 0; u < 4; ++u) {
    int d = lane * 4 + u;
    float xn = v[u] * rs * qw[d];
    float partner = __shfl_xor(xn, 16);        // value at d ^ 64
    float rot = (lane < 16) ? -partner : partner;
    float ang = pos * __powf(10000.0f, -(float)(d & 63) / 64.0f);
    dst[d] = (_Float16)(xn * __cosf(ang) + rot * __sinf(ang));
  }
}

// ---------------------------------------------------------------------------
// RMSNorm K in place (per (s,kv) row of 128).
// ---------------------------------------------------------------------------
__global__ void __launch_bounds__(128) rms_k_kernel(float* __restrict__ k,
                                                    const float* __restrict__ kw) {
  int wave = threadIdx.x >> 5, lane = threadIdx.x & 31;
  int rowid = blockIdx.x * 4 + wave;           // s*NKV + kv
  int s = rowid / NKV, kv = rowid % NKV;
  float* row = k + (size_t)s * (NKV * HD) + (size_t)kv * HD;
  float v[4];
  float ss = 0.0f;
#pragma unroll
  for (int u = 0; u < 4; ++u) { v[u] = row[lane * 4 + u]; ss += v[u] * v[u]; }
  for (int off = 16; off > 0; off >>= 1) ss += __shfl_xor(ss, off);
  float rs = rsqrtf(ss / (float)HD + EPSV);
#pragma unroll
  for (int u = 0; u < 4; ++u) {
    int d = lane * 4 + u;
    row[d] = v[u] * rs * kw[d];
  }
}

// ---------------------------------------------------------------------------
// Gather top-K keys/values: RoPE gathered K -> kh[kv][t][d]; V -> vt[kv][d][t].
// vis[t] = position (visibility time).
// ---------------------------------------------------------------------------
__global__ void __launch_bounds__(128) gather_topk_kernel(const float* __restrict__ kn,
                                                          const float* __restrict__ v,
                                                          const int* __restrict__ idx,
                                                          const int* __restrict__ pos_ids,
                                                          _Float16* __restrict__ kh,
                                                          _Float16* __restrict__ vt,
                                                          float* __restrict__ vis) {
  int t = blockIdx.x;
  int d = threadIdx.x;                          // 0..127
  int s = idx[t];
  if (d == 0) vis[t] = (float)s;
  float pos = (float)pos_ids[s];
  float ang = pos * __powf(10000.0f, -(float)(d & 63) / 64.0f);
  float c = __cosf(ang), si = __sinf(ang);
  for (int kv = 0; kv < NKV; ++kv) {
    const float* kr = kn + (size_t)s * (NKV * HD) + (size_t)kv * HD;
    float x0 = kr[d];
    float xp = (d < 64) ? kr[d + 64] : kr[d - 64];
    float rot = (d < 64) ? -xp : xp;
    kh[((size_t)kv * T_TOT + t) * HD + d] = (_Float16)(x0 * c + rot * si);
    vt[((size_t)kv * HD + d) * T_TOT + t] =
        (_Float16)v[(size_t)s * (NKV * HD) + (size_t)kv * HD + d];
  }
}

// ---------------------------------------------------------------------------
// Kron-DCT sketch of K (tensor=0, uses rms-normed K) and V (tensor=1).
// One block per (tensor, kv, d). Writes sketch rows into kh / vt + vis times.
// ---------------------------------------------------------------------------
__global__ void __launch_bounds__(256) sketch_kernel(const float* __restrict__ kn,
                                                     const float* __restrict__ vf,
                                                     const float* __restrict__ restw,
                                                     const float* __restrict__ kron_a,
                                                     const float* __restrict__ kron_b,
                                                     const float* __restrict__ sscale,
                                                     _Float16* __restrict__ kh,
                                                     _Float16* __restrict__ vt,
                                                     float* __restrict__ vis) {
  int bid = blockIdx.x;
  int tensor = bid >> 9;
  int kv = (bid >> 7) & 3;
  int d = bid & 127;
  const float* src = tensor ? vf : kn;
  __shared__ float w[S_LEN];
  __shared__ float y1[16][32];
  int tid = threadIdx.x;
  for (int s = tid; s < S_LEN; s += 256)
    w[s] = src[(size_t)s * (NKV * HD) + (size_t)kv * HD + d] * restw[s];
  __syncthreads();
  // y1[p][c] = sum_q w[p*256+q] * cb[c][q]; mask: q < (c+1)*8; only p<16 nonzero
  for (int e = tid; e < 512; e += 256) {
    int p = e >> 5, c = e & 31;
    int qlim = (c + 1) * 8;
    float sum = 0.0f;
    for (int q2 = 0; q2 < qlim; ++q2) sum = fmaf(w[p * 256 + q2], kron_b[c * 256 + q2], sum);
    y1[p][c] = sum;
  }
  __syncthreads();
  float scale = sscale[0];
  // y2[c][a] = sum_p y1[p][c] * ca[a][p]; mask: p < floor((a+1)*128/20)
  for (int e = tid; e < SK; e += 256) {
    int c = e / 20, a = e % 20;
    int plim = ((a + 1) * 32) / 5;
    if (plim > 16) plim = 16;                  // p>=16 rows are zero padding
    float sum = 0.0f;
    for (int p = 0; p < plim; ++p) sum = fmaf(y1[p][c], kron_a[a * 128 + p], sum);
    sum *= scale;
    int j = c * 20 + a;
    if (tensor == 0) kh[((size_t)kv * T_TOT + (KTOP + j)) * HD + d] = (_Float16)sum;
    else             vt[((size_t)kv * HD + d) * T_TOT + (KTOP + j)] = (_Float16)sum;
  }
  if (tensor == 0 && kv == 0 && d == 0)
    for (int j = tid; j < SK; j += 256)
      vis[KTOP + j] = (float)j * 51.2f;        // (KA_C*KB_C)/SKETCH
}

// ---------------------------------------------------------------------------
// Flash attention, one wave per (head, 16-query block). WMMA QK^T and P*V,
// LDS-staged online softmax, visibility masking via vis[] times.
// ---------------------------------------------------------------------------
__global__ void __launch_bounds__(32) attn_wmma_kernel(const _Float16* __restrict__ qh,
                                                       const _Float16* __restrict__ kh,
                                                       const _Float16* __restrict__ vt,
                                                       const float* __restrict__ vis,
                                                       float* __restrict__ ao) {
  __builtin_amdgcn_s_wait_tensorcnt(0);
  int h = blockIdx.x, qb = blockIdx.y;
  int lane = threadIdx.x;
  int kv = h / (NH / NKV);
  const _Float16* qbase = qh + ((size_t)h * S_LEN + (size_t)qb * 16) * HD;
  const _Float16* kbase = kh + (size_t)kv * T_TOT * HD;
  const _Float16* vbase = vt + (size_t)kv * HD * T_TOT;
  __shared__ float sm[16][32];
  __shared__ _Float16 pt[16][32];
  __shared__ float rowf[16];
  __shared__ float rowl[16];
  v16h qa[4];
#pragma unroll
  for (int i = 0; i < 4; ++i) qa[i] = load_a_frag(qbase, 0, i * 32, HD);
  v8f acc[8] = {};
  float row_m = -INFINITY, row_l = 0.0f;
  float qf = (float)(qb * 16 + (lane & 15));
  const float scale = 0.08838834764831845f;    // 1/sqrt(128)
  for (int t0 = 0; t0 < T_TOT; t0 += 32) {
    v8f s0 = {}, s1 = {};
#pragma unroll
    for (int i = 0; i < 4; ++i) {
      s0 = __builtin_amdgcn_wmma_f32_16x16x32_f16(false, qa[i], false,
             load_b_frag(kbase, t0, i * 32, HD), (short)0, s0, false, false);
      s1 = __builtin_amdgcn_wmma_f32_16x16x32_f16(false, qa[i], false,
             load_b_frag(kbase, t0 + 16, i * 32, HD), (short)0, s1, false, false);
    }
    int col = lane & 15;
#pragma unroll
    for (int i = 0; i < 8; ++i) {
      int row = (lane & 16) ? i + 8 : i;
      sm[row][col]      = s0[i] * scale;
      sm[row][col + 16] = s1[i] * scale;
    }
    __syncthreads();
    if (lane < 16) {
      float sv[32];
      float cmax = -INFINITY;
#pragma unroll
      for (int j = 0; j < 32; ++j) {
        float x = sm[lane][j];
        if (vis[t0 + j] > qf) x = -INFINITY;   // not yet visible to this query
        sv[j] = x;
        cmax = fmaxf(cmax, x);
      }
      float mnew = fmaxf(row_m, cmax);
      float f, sum = 0.0f;
      if (mnew == -INFINITY) {
        f = 1.0f;
#pragma unroll
        for (int j = 0; j < 32; ++j) pt[lane][j] = (_Float16)0.0f;
      } else {
        f = __expf(row_m - mnew);
#pragma unroll
        for (int j = 0; j < 32; ++j) {
          float p = (sv[j] == -INFINITY) ? 0.0f : __expf(sv[j] - mnew);
          sum += p;
          pt[lane][j] = (_Float16)p;
        }
      }
      row_l = row_l * f + sum;
      row_m = mnew;
      rowf[lane] = f;
    }
    __syncthreads();
#pragma unroll
    for (int t = 0; t < 8; ++t)
#pragma unroll
      for (int i = 0; i < 8; ++i) {
        int row = (lane & 16) ? i + 8 : i;
        acc[t][i] *= rowf[row];
      }
    v16h pa = load_a_frag(&pt[0][0], 0, 0, 32);
#pragma unroll
    for (int t = 0; t < 8; ++t)
      acc[t] = __builtin_amdgcn_wmma_f32_16x16x32_f16(false, pa, false,
                 load_b_frag(vbase, t * 16, t0, T_TOT), (short)0, acc[t], false, false);
    __syncthreads();
  }
  if (lane < 16) rowl[lane] = (row_l > 0.0f) ? 1.0f / row_l : 0.0f;
  __syncthreads();
  int col = lane & 15;
#pragma unroll
  for (int t = 0; t < 8; ++t)
#pragma unroll
    for (int i = 0; i < 8; ++i) {
      int row = (lane & 16) ? i + 8 : i;
      ao[(size_t)(qb * 16 + row) * (NH * HD) + (size_t)h * HD + t * 16 + col] =
          acc[t][i] * rowl[row];
    }
}

// ---------------------------------------------------------------------------
// Host-side orchestration
// ---------------------------------------------------------------------------
extern "C" void kernel_launch(void* const* d_in, const int* in_sizes, int n_in,
                              void* d_out, int out_size, void* d_ws, size_t ws_size,
                              hipStream_t stream) {
  (void)in_sizes; (void)n_in; (void)out_size; (void)ws_size;
  const float* x       = (const float*)d_in[0];
  const int*   pos     = (const int*)d_in[1];
  const float* Wq      = (const float*)d_in[2];
  const float* Wk      = (const float*)d_in[3];
  const float* Wv      = (const float*)d_in[4];
  const float* Wo      = (const float*)d_in[5];
  const float* qnw     = (const float*)d_in[6];
  const float* knw     = (const float*)d_in[7];
  const float* kron_a  = (const float*)d_in[8];
  const float* kron_b  = (const float*)d_in[9];
  const float* sscale  = (const float*)d_in[10];
  const float* W1      = (const float*)d_in[11];
  const float* b1      = (const float*)d_in[12];
  const float* W2      = (const float*)d_in[13];
  const float* b2      = (const float*)d_in[14];
  float* out = (float*)d_out;

  char* ws = (char*)d_ws;
  size_t off = 0;
  auto alloc = [&](size_t bytes) -> void* {
    void* p = (void*)(ws + off);
    off += (bytes + 255) & ~(size_t)255;
    return p;
  };
  _Float16* xh   = (_Float16*)alloc((size_t)S_LEN * HIDN * 2);
  _Float16* wqT  = (_Float16*)alloc((size_t)HIDN * HIDN * 2);
  _Float16* wkT  = (_Float16*)alloc((size_t)(NKV * HD) * HIDN * 2);
  _Float16* wvT  = (_Float16*)alloc((size_t)(NKV * HD) * HIDN * 2);
  _Float16* woT  = (_Float16*)alloc((size_t)HIDN * HIDN * 2);
  float* qf      = (float*)alloc((size_t)S_LEN * (NH * HD) * 4);
  float* kf      = (float*)alloc((size_t)S_LEN * (NKV * HD) * 4);
  float* vf      = (float*)alloc((size_t)S_LEN * (NKV * HD) * 4);
  float* logits  = (float*)alloc((size_t)S_LEN * 4);
  int*   idx     = (int*)alloc((size_t)KTOP * 4);
  int*   istop   = (int*)alloc((size_t)S_LEN * 4);
  float* restw   = (float*)alloc((size_t)S_LEN * 4);
  float* vis     = (float*)alloc((size_t)T_TOT * 4);
  _Float16* qh   = (_Float16*)alloc((size_t)NH * S_LEN * HD * 2);
  _Float16* kh   = (_Float16*)alloc((size_t)NKV * T_TOT * HD * 2);
  _Float16* vt   = (_Float16*)alloc((size_t)NKV * HD * T_TOT * 2);
  float* ao      = (float*)alloc((size_t)S_LEN * (NH * HD) * 4);
  _Float16* aoh  = (_Float16*)alloc((size_t)S_LEN * (NH * HD) * 2);

  // 1) f16 copies / transposed weights
  f32_to_f16_kernel<<<(S_LEN * HIDN) / 256, 256, 0, stream>>>(x, xh, S_LEN * HIDN);
  transpose_w_kernel<<<dim3(HIDN / 32, HIDN / 32), dim3(32, 8), 0, stream>>>(Wq, wqT, HIDN, HIDN);
  transpose_w_kernel<<<dim3((NKV * HD) / 32, HIDN / 32), dim3(32, 8), 0, stream>>>(Wk, wkT, HIDN, NKV * HD);
  transpose_w_kernel<<<dim3((NKV * HD) / 32, HIDN / 32), dim3(32, 8), 0, stream>>>(Wv, wvT, HIDN, NKV * HD);
  transpose_w_kernel<<<dim3(HIDN / 32, HIDN / 32), dim3(32, 8), 0, stream>>>(Wo, woT, HIDN, HIDN);

  // 2) projections (WMMA, 64x64 register tiles, block tile 64x256)
  gemm_f16_wmma<<<dim3(HIDN / 256, S_LEN / 64), 128, 0, stream>>>(xh, wqT, qf, S_LEN, HIDN, HIDN);
  gemm_f16_wmma<<<dim3((NKV * HD) / 256, S_LEN / 64), 128, 0, stream>>>(xh, wkT, kf, S_LEN, NKV * HD, HIDN);
  gemm_f16_wmma<<<dim3((NKV * HD) / 256, S_LEN / 64), 128, 0, stream>>>(xh, wvT, vf, S_LEN, NKV * HD, HIDN);

  // 3) importance scores + exact top-k + residual weights
  mlp_logits_kernel<<<S_LEN, 64, 0, stream>>>(x, W1, b1, W2, b2, logits);
  topk_select_kernel<<<1, 1024, 0, stream>>>(logits, idx, istop);
  restw_kernel<<<S_LEN / 256, 256, 0, stream>>>(logits, istop, restw);

  // 4) norms + rope; build attention-side K/V
  rms_rope_q_kernel<<<(S_LEN * NH) / 4, 128, 0, stream>>>(qf, pos, qnw, qh);
  rms_k_kernel<<<(S_LEN * NKV) / 4, 128, 0, stream>>>(kf, knw);
  gather_topk_kernel<<<KTOP, 128, 0, stream>>>(kf, vf, idx, pos, kh, vt, vis);
  sketch_kernel<<<2 * NKV * HD, 256, 0, stream>>>(kf, vf, restw, kron_a, kron_b, sscale,
                                                  kh, vt, vis);

  // 5) flash attention (WMMA)
  attn_wmma_kernel<<<dim3(NH, S_LEN / 16), 32, 0, stream>>>(qh, kh, vt, vis, ao);

  // 6) output projection (WMMA)
  f32_to_f16_kernel<<<(S_LEN * HIDN) / 256, 256, 0, stream>>>(ao, aoh, S_LEN * HIDN);
  gemm_f16_wmma<<<dim3(HIDN / 256, S_LEN / 64), 128, 0, stream>>>(aoh, woT, out, S_LEN, HIDN, HIDN);
}
